// TemporalEncoder_51745765982729
// MI455X (gfx1250) — matrix operand order repaired and checked by reading, not compiled, not run on hardware
//
#include <hip/hip_runtime.h>
#include <hip/hip_bf16.h>
#include <stdint.h>

#define Bz 64
#define Tz 1024
#define Hz 256
#define NWG 8              // workgroups cooperating per direction (split 4H across them)
#define THREADS 256        // 8 waves of 32

typedef __attribute__((ext_vector_type(16))) _Float16 v16h;
typedef __attribute__((ext_vector_type(8)))  _Float16 v8h;
typedef __attribute__((ext_vector_type(8)))  float    v8f;
typedef __attribute__((ext_vector_type(4)))  unsigned int v4u;
typedef __attribute__((ext_vector_type(8)))  int      v8i;
typedef __attribute__((ext_vector_type(4)))  int      v4i;

#if defined(__gfx1250__) && __has_builtin(__builtin_amdgcn_tensor_load_to_lds)
#define USE_TDM 1
#else
#define USE_TDM 0
#endif

// ---------------------------------------------------------------------------
// Stage a 2D tile (rows x row_elems of 2-byte elements, row pitch
// row_stride_elems) from global memory into LDS, packed contiguously.
// gfx1250 path: Tensor Data Mover (async DMA, tracked with TENSORcnt),
// 6-arg builtin: (uint32x4 g0, int32x8 g1, int32x4, int32x4, int32x8, i32).
// ---------------------------------------------------------------------------
__device__ __forceinline__ void stage_tile(_Float16* lds_ptr, uint32_t lds_byte_off,
                                           const _Float16* gsrc, uint32_t row_elems,
                                           uint32_t rows, uint64_t row_stride_elems)
{
#if USE_TDM
    if (threadIdx.x < 32) {   // one wave issues the DMA
        uint64_t ga = (uint64_t)(uintptr_t)gsrc;
        v4u g0;
        g0.x = 1u;                                              // count=1 valid descriptor
        g0.y = lds_byte_off;                                    // lds_addr
        g0.z = (uint32_t)ga;                                    // global_addr[31:0]
        g0.w = (uint32_t)((ga >> 32) & 0x01ffffffull) | (2u << 30); // global_addr[56:32] | type=2
        v8i g1;
        g1[0] = (int)(1u << 16);                                // data_size = 2 bytes
        g1[1] = (int)((row_elems & 0xffffu) << 16);             // tensor_dim0[15:0]
        g1[2] = (int)(((row_elems >> 16) & 0xffffu) |
                      ((rows & 0xffffu) << 16));                // tensor_dim0[31:16] | tensor_dim1[15:0]
        g1[3] = (int)(((rows >> 16) & 0xffffu) |
                      ((row_elems & 0xffffu) << 16));           // tensor_dim1[31:16] | tile_dim0
        g1[4] = (int)(rows & 0xffffu);                          // tile_dim1 (tile_dim2 = 0)
        g1[5] = (int)(uint32_t)row_stride_elems;                // tensor_dim0_stride[31:0]
        g1[6] = (int)(uint32_t)((row_stride_elems >> 32) & 0xffffull); // stride[47:32]
        g1[7] = 0;
        v4i z4 = {0, 0, 0, 0};
        v8i z8 = {0, 0, 0, 0, 0, 0, 0, 0};
        __builtin_amdgcn_tensor_load_to_lds(g0, g1, z4, z4, z8, 0);
    }
    (void)lds_ptr;
#else
    (void)lds_byte_off;
    for (uint32_t i = threadIdx.x; i < rows * row_elems; i += blockDim.x) {
        uint32_t r = i / row_elems, c = i % row_elems;
        lds_ptr[(size_t)r * row_elems + c] = gsrc[(size_t)r * row_stride_elems + c];
    }
#endif
}

__device__ __forceinline__ void stage_wait()
{
#if USE_TDM
    if (threadIdx.x < 32) __builtin_amdgcn_s_wait_tensorcnt(0);
#endif
    __syncthreads();
}

__device__ __forceinline__ float sigmoidf_(float x) { return 1.0f / (1.0f + __expf(-x)); }

__device__ __forceinline__ v8f splat8(float v)
{
    v8f r;
#pragma unroll
    for (int i = 0; i < 8; ++i) r[i] = v;
    return r;
}

__device__ __forceinline__ v16h load_a_tile(const _Float16* ap, int lhi)
{
    v8h alo = *(const v8h*)(ap + lhi * 16);
    v8h ahi = *(const v8h*)(ap + lhi * 16 + 8);
    return __builtin_shufflevector(alo, ahi, 0,1,2,3,4,5,6,7,8,9,10,11,12,13,14,15);
}

// ---------------------------------------------------------------------------
// Prep kernel: f32->f16 conversion of x, fused+reordered f16 weight blocks,
// summed reordered biases, and zeroing of the per-step barrier counters.
// Reorder: n' = unit_tile*64 + gate*16 + unit_lo so each wave owns i,f,g,o
// of one 16-unit slice (cell state stays in registers).
// ---------------------------------------------------------------------------
__global__ void lstm_prep_kernel(const float* __restrict__ x,
                                 const float* __restrict__ w_ih0, const float* __restrict__ w_hh0,
                                 const float* __restrict__ b_ih0, const float* __restrict__ b_hh0,
                                 const float* __restrict__ w_ih1, const float* __restrict__ w_hh1,
                                 const float* __restrict__ b_ih1, const float* __restrict__ b_hh1,
                                 _Float16* __restrict__ x16,
                                 _Float16* __restrict__ Wcat0, _Float16* __restrict__ Wcat1,
                                 float* __restrict__ bcat0, float* __restrict__ bcat1,
                                 int* __restrict__ counters)
{
    const size_t NX  = (size_t)Bz * Tz * Hz;      // x elements
    const size_t NW0 = (size_t)2 * 1024 * 512;    // layer0 fused weights
    const size_t NW1 = (size_t)2 * 1024 * 768;    // layer1 fused weights
    const size_t NB  = 2 * 1024;                  // biases per layer
    const size_t NC  = 2 * 2 * Tz;                // barrier counters (2 layers x 2 dirs x T)
    const size_t TOT = NX + NW0 + NW1 + 2 * NB + NC;
    const size_t stride = (size_t)gridDim.x * blockDim.x;

    for (size_t idx = (size_t)blockIdx.x * blockDim.x + threadIdx.x; idx < TOT; idx += stride) {
        if (idx < NX) {
            x16[idx] = (_Float16)x[idx];
        } else if (idx < NX + NW0) {
            size_t j = idx - NX;
            int k = (int)(j % 512);
            size_t nj = j / 512;
            int np = (int)(nj % 1024), d = (int)(nj / 1024);
            int utile = np >> 6, gate = (np >> 4) & 3, ul = np & 15;
            int n = gate * 256 + utile * 16 + ul;
            float v = (k < 256) ? w_hh0[((size_t)d * 1024 + n) * 256 + k]
                                : w_ih0[((size_t)d * 1024 + n) * 256 + (k - 256)];
            Wcat0[j] = (_Float16)v;
        } else if (idx < NX + NW0 + NW1) {
            size_t j = idx - NX - NW0;
            int k = (int)(j % 768);
            size_t nj = j / 768;
            int np = (int)(nj % 1024), d = (int)(nj / 1024);
            int utile = np >> 6, gate = (np >> 4) & 3, ul = np & 15;
            int n = gate * 256 + utile * 16 + ul;
            float v = (k < 256) ? w_hh1[((size_t)d * 1024 + n) * 256 + k]
                                : w_ih1[((size_t)d * 1024 + n) * 512 + (k - 256)];
            Wcat1[j] = (_Float16)v;
        } else if (idx < NX + NW0 + NW1 + NB) {
            size_t j = idx - NX - NW0 - NW1;
            int np = (int)(j % 1024), d = (int)(j / 1024);
            int utile = np >> 6, gate = (np >> 4) & 3, ul = np & 15;
            int n = gate * 256 + utile * 16 + ul;
            bcat0[j] = b_ih0[(size_t)d * 1024 + n] + b_hh0[(size_t)d * 1024 + n];
        } else if (idx < NX + NW0 + NW1 + 2 * NB) {
            size_t j = idx - NX - NW0 - NW1 - NB;
            int np = (int)(j % 1024), d = (int)(j / 1024);
            int utile = np >> 6, gate = (np >> 4) & 3, ul = np & 15;
            int n = gate * 256 + utile * 16 + ul;
            bcat1[j] = b_ih1[(size_t)d * 1024 + n] + b_hh1[(size_t)d * 1024 + n];
        } else {
            counters[idx - (NX + NW0 + NW1 + 2 * NB)] = 0;
        }
    }
}

// ---------------------------------------------------------------------------
// Recurrent scan: grid = (NWG, 2 dirs). Each workgroup owns a 128-row slice
// of the reordered fused weights in LDS (TDM-staged once) and 2 unit-tiles
// of the hidden state. Per step:
//   TDM-stage x_t and h_t -> LDS, wait TENSORcnt,
//   gates = [h|x] @ Wslice^T + b via v_wmma_f32_16x16x32_f16,
//   c,h update in registers, h -> global double-buffered exchange,
//   device-scope atomic barrier over the NWG WGPs of this direction.
// Templated on In so all LDS strides are compile-time immediates.
// ---------------------------------------------------------------------------
template <int IN>
__global__ __launch_bounds__(THREADS)
void lstm_scan_kernel(const _Float16* __restrict__ Wcat,  // [2][1024][K] reordered f16
                      const float* __restrict__ bcat,     // [2][1024] reordered
                      const _Float16* __restrict__ xin,   // [B][T][IN] f16
                      void* __restrict__ outp, int outIsF16, // [B][T][512]
                      _Float16* __restrict__ hbuf,        // [2 dirs][2 phases][64][256] f16
                      int* __restrict__ cnt,              // [2 dirs][T] barrier counters
                      float* __restrict__ hiddenOut,      // [4][64][256]
                      float* __restrict__ cellOut,        // [4][64][256]
                      int layer)
{
    extern __shared__ char smem[];
    constexpr int K  = 256 + IN;
    constexpr int XB = IN / 32;                 // x-segment K-blocks
    const int dir = blockIdx.y;
    const int wg  = blockIdx.x;

    _Float16* sW = (_Float16*)smem;                           // [128][K]
    _Float16* sH = (_Float16*)(smem + (size_t)128 * K * 2);   // [64][256]
    _Float16* sX = sH + 64 * 256;                             // [64][IN]
    constexpr uint32_t HA_OFF = 128u * (uint32_t)K * 2u;
    constexpr uint32_t XA_OFF = HA_OFF + 64u * 256u * 2u;

    const int tid  = threadIdx.x;
    const int lane = tid & 31, wave = tid >> 5;
    const int mt = wave & 3;        // batch tile 0..3 (16 rows each)
    const int ut = wave >> 2;       // unit tile within WGP 0..1
    const int lcol = lane & 15, lhi = lane >> 4;

    // ---- stage this WGP's weight slice (once) via TDM ----
    const _Float16* Wslice = Wcat + (size_t)dir * 1024 * K + (size_t)wg * 128 * K;
    stage_tile(sW, 0, Wslice, (uint32_t)K, 128, (uint64_t)K);
    stage_wait();

    // per-wave bias for its 4 gate columns
    float bi[4];
#pragma unroll
    for (int g = 0; g < 4; ++g)
        bi[g] = bcat[(size_t)dir * 1024 + wg * 128 + ut * 64 + g * 16 + lcol];

    const int arow = mt * 16 + lcol;            // A-matrix lane row
    const int ucol = wg * 32 + ut * 16 + lcol;  // global hidden unit of this lane column
    const _Float16* aH = sH + (size_t)arow * 256;
    const _Float16* aX = sX + (size_t)arow * IN;
    const _Float16* bW = sW + (size_t)(ut * 64 + lcol) * K + lhi * 16;
    _Float16* hbdir = hbuf + (size_t)dir * 2 * 64 * 256;
    int* cdir = cnt + (size_t)dir * Tz;

    v8f cst = {};                               // cell state (16x16 tile, in registers)

    for (int s = 0; s < Tz; ++s) {
        const int tt = dir ? (Tz - 1 - s) : s;

        // ---- stage x_t and h_s into LDS ----
        stage_tile(sX, XA_OFF, xin + (size_t)tt * IN, (uint32_t)IN, 64, (uint64_t)Tz * IN);
        if (s == 0) {
            for (int i = tid; i < 64 * 256; i += THREADS) sH[i] = (_Float16)0.0f;
        } else {
            stage_tile(sH, HA_OFF, hbdir + (size_t)(s & 1) * 64 * 256, 256, 64, 256);
        }
        stage_wait();

        // ---- gates = [h | x] @ W^T + b  (f16 WMMA, f32 accumulate) ----
        v8f acc0 = splat8(bi[0]);
        v8f acc1 = splat8(bi[1]);
        v8f acc2 = splat8(bi[2]);
        v8f acc3 = splat8(bi[3]);

#pragma unroll
        for (int kb = 0; kb < 8; ++kb) {        // recurrent segment: K 0..255 from sH
            v16h a  = load_a_tile(aH + kb * 32, lhi);
            v16h b0 = *(const v16h*)(bW + kb * 32 + (size_t)0 * 16 * K);
            v16h b1 = *(const v16h*)(bW + kb * 32 + (size_t)1 * 16 * K);
            v16h b2 = *(const v16h*)(bW + kb * 32 + (size_t)2 * 16 * K);
            v16h b3 = *(const v16h*)(bW + kb * 32 + (size_t)3 * 16 * K);
            acc0 = __builtin_amdgcn_wmma_f32_16x16x32_f16(false, a, false, b0, (short)0, acc0, false, false);
            acc1 = __builtin_amdgcn_wmma_f32_16x16x32_f16(false, a, false, b1, (short)0, acc1, false, false);
            acc2 = __builtin_amdgcn_wmma_f32_16x16x32_f16(false, a, false, b2, (short)0, acc2, false, false);
            acc3 = __builtin_amdgcn_wmma_f32_16x16x32_f16(false, a, false, b3, (short)0, acc3, false, false);
        }
#pragma unroll
        for (int kb = 0; kb < XB; ++kb) {       // input segment: K 256.. from sX
            v16h a  = load_a_tile(aX + kb * 32, lhi);
            const _Float16* bp = bW + 256 + kb * 32;
            v16h b0 = *(const v16h*)(bp + (size_t)0 * 16 * K);
            v16h b1 = *(const v16h*)(bp + (size_t)1 * 16 * K);
            v16h b2 = *(const v16h*)(bp + (size_t)2 * 16 * K);
            v16h b3 = *(const v16h*)(bp + (size_t)3 * 16 * K);
            acc0 = __builtin_amdgcn_wmma_f32_16x16x32_f16(false, a, false, b0, (short)0, acc0, false, false);
            acc1 = __builtin_amdgcn_wmma_f32_16x16x32_f16(false, a, false, b1, (short)0, acc1, false, false);
            acc2 = __builtin_amdgcn_wmma_f32_16x16x32_f16(false, a, false, b2, (short)0, acc2, false, false);
            acc3 = __builtin_amdgcn_wmma_f32_16x16x32_f16(false, a, false, b3, (short)0, acc3, false, false);
        }

        // ---- LSTM cell update (register-local) ----
        v8f hv;
#pragma unroll
        for (int r = 0; r < 8; ++r) {
            float iv = sigmoidf_(acc0[r]);
            float fv = sigmoidf_(acc1[r]);
            float gv = tanhf(acc2[r]);
            float ov = sigmoidf_(acc3[r]);
            float c = fv * cst[r] + iv * gv;
            cst[r] = c;
            hv[r] = ov * tanhf(c);
        }

        // ---- write h to exchange buffer + sequence output ----
        _Float16* hbn = hbdir + (size_t)((s + 1) & 1) * 64 * 256;
#pragma unroll
        for (int r = 0; r < 8; ++r) {
            int m = mt * 16 + lhi * 8 + r;
            hbn[(size_t)m * 256 + ucol] = (_Float16)hv[r];
            size_t oidx = ((size_t)m * Tz + tt) * 512 + (size_t)dir * 256 + ucol;
            if (outIsF16) ((_Float16*)outp)[oidx] = (_Float16)hv[r];
            else          ((float*)outp)[oidx]    = hv[r];
        }
        if (s == Tz - 1) {
#pragma unroll
            for (int r = 0; r < 8; ++r) {
                int m = mt * 16 + lhi * 8 + r;
                size_t hidx = ((size_t)(layer * 2 + dir) * 64 + m) * 256 + ucol;
                hiddenOut[hidx] = hv[r];
                cellOut[hidx]   = cst[r];
            }
        }

        // ---- device-scope barrier over the NWG workgroups of this direction ----
        __threadfence();
        __syncthreads();
        if (tid == 0) {
            int* cp = cdir + s;
            __hip_atomic_fetch_add(cp, 1, __ATOMIC_RELEASE, __HIP_MEMORY_SCOPE_AGENT);
            while (__hip_atomic_load(cp, __ATOMIC_ACQUIRE, __HIP_MEMORY_SCOPE_AGENT) < NWG) {
#if defined(__HIP_DEVICE_COMPILE__)
                __builtin_amdgcn_s_sleep(1);
#endif
            }
        }
        __syncthreads();
    }
}

// ---------------------------------------------------------------------------
// Host launch
// ---------------------------------------------------------------------------
extern "C" void kernel_launch(void* const* d_in, const int* in_sizes, int n_in,
                              void* d_out, int out_size, void* d_ws, size_t ws_size,
                              hipStream_t stream)
{
    const float* x     = (const float*)d_in[0];
    const float* w_ih0 = (const float*)d_in[1];
    const float* w_hh0 = (const float*)d_in[2];
    const float* b_ih0 = (const float*)d_in[3];
    const float* b_hh0 = (const float*)d_in[4];
    const float* w_ih1 = (const float*)d_in[5];
    const float* w_hh1 = (const float*)d_in[6];
    const float* b_ih1 = (const float*)d_in[7];
    const float* b_hh1 = (const float*)d_in[8];
    (void)in_sizes; (void)n_in; (void)out_size; (void)ws_size;

    // workspace layout (bytes)
    char* ws = (char*)d_ws;
    size_t off = 0;
    _Float16* o016  = (_Float16*)(ws + off); off += (size_t)Bz * Tz * 512 * 2;   // 64 MB
    _Float16* x16   = (_Float16*)(ws + off); off += (size_t)Bz * Tz * 256 * 2;   // 32 MB
    _Float16* Wcat0 = (_Float16*)(ws + off); off += (size_t)2 * 1024 * 512 * 2;  // 2 MB
    _Float16* Wcat1 = (_Float16*)(ws + off); off += (size_t)2 * 1024 * 768 * 2;  // 3 MB
    float*    bcat0 = (float*)(ws + off);    off += (size_t)2 * 1024 * 4;
    float*    bcat1 = (float*)(ws + off);    off += (size_t)2 * 1024 * 4;
    _Float16* hbuf  = (_Float16*)(ws + off); off += (size_t)2 * 2 * 64 * 256 * 2;
    int*      cnt0  = (int*)(ws + off);      off += (size_t)2 * Tz * 4;
    int*      cnt1  = (int*)(ws + off);      off += (size_t)2 * Tz * 4;
    int*      cntall = cnt0;  // prep zeroes cnt0 and cnt1 contiguously

    float* out_f   = (float*)d_out;
    float* hiddenP = out_f + (size_t)Bz * Tz * 512;
    float* cellP   = hiddenP + (size_t)4 * 64 * 256;

    lstm_prep_kernel<<<2048, 256, 0, stream>>>(x, w_ih0, w_hh0, b_ih0, b_hh0,
                                               w_ih1, w_hh1, b_ih1, b_hh1,
                                               x16, Wcat0, Wcat1, bcat0, bcat1, cntall);

    const size_t shmem0 = (size_t)128 * 512 * 2 + 64 * 256 * 2 + 64 * 256 * 2; // 192 KB
    const size_t shmem1 = (size_t)128 * 768 * 2 + 64 * 256 * 2 + 64 * 512 * 2; // 288 KB
    (void)hipFuncSetAttribute((const void*)lstm_scan_kernel<256>,
                              hipFuncAttributeMaxDynamicSharedMemorySize, (int)shmem0);
    (void)hipFuncSetAttribute((const void*)lstm_scan_kernel<512>,
                              hipFuncAttributeMaxDynamicSharedMemorySize, (int)shmem1);

    // layer 0: input x_f16 (In=256) -> o0 (f16)
    lstm_scan_kernel<256><<<dim3(NWG, 2), THREADS, shmem0, stream>>>(
        Wcat0, bcat0, x16, (void*)o016, 1, hbuf, cnt0, hiddenP, cellP, 0);

    // layer 1: input o0_f16 (In=512) -> o1 (f32, d_out)
    lstm_scan_kernel<512><<<dim3(NWG, 2), THREADS, shmem1, stream>>>(
        Wcat1, bcat1, o016, (void*)out_f, 0, hbuf, cnt1, hiddenP, cellP, 1);
}
